// C2VQ_Recon_Net_48885317763881
// MI455X (gfx1250) — compile-verified, matrix operand order checked by reading, not run on hardware
//
#include <hip/hip_runtime.h>
#include <math.h>

// ---------------- types ----------------
typedef _Float16 h_t;
typedef __attribute__((ext_vector_type(16))) _Float16 v16h;
typedef __attribute__((ext_vector_type(8)))  _Float16 h8;
typedef __attribute__((ext_vector_type(4)))  _Float16 h4;
typedef __attribute__((ext_vector_type(2)))  _Float16 h2;
typedef __attribute__((ext_vector_type(8)))  float    v8f;
typedef unsigned int u32x4 __attribute__((ext_vector_type(4)));
typedef int          i32x8 __attribute__((ext_vector_type(8)));
typedef int          i32x4 __attribute__((ext_vector_type(4)));

#define N_IMG 512
#define CD    256
#define DDIM  4096
#define KCODE 2048
#define NCLS  8

// Load a 16-half A/B fragment from an LDS row laid out [row][40] (stride 40 halves).
// kb = (lane>>4)*8 per the CDNA5 16-bit operand VGPR layout:
//   lanes 0-15:  halves 0..7 = K 0..7,  halves 8..15 = K 16..23
//   lanes 16-31: halves 0..7 = K 8..15, halves 8..15 = K 24..31
__device__ inline v16h frag_ld(const h_t* rowbase, int kb) {
  h8 lo = *(const h8*)(rowbase + kb);
  h8 hi = *(const h8*)(rowbase + kb + 16);
  return __builtin_shufflevector(lo, hi, 0,1,2,3,4,5,6,7,8,9,10,11,12,13,14,15);
}

__device__ inline v8f wmma16(v16h a, v16h b, v8f c) {
  return __builtin_amdgcn_wmma_f32_16x16x32_f16(false, a, false, b, (short)0, c, false, false);
}

// ---- Tensor Data Mover: 2D tile (t0 halves x t1 rows, row stride stride0 halves)
// from global into LDS laid out [t1][40] halves (64B data + 16B pad per row).
// D# per CDNA5 ISA:
//   group0: [1:0]count=1 | [63:32]lds_addr | [120:64]global_addr | [127:126]type=2
//   group1: [17:16]data_size=1(2B) [20]pad_en [24:22]pad_interval=3(16DW) [31:25]pad_amount=3(4DW)
//           [79:48]tensor_dim0 [111:80]tensor_dim1 [127:112]tile_dim0 [143:128]tile_dim1
//           [207:160]tensor_dim0_stride
__device__ inline void tdm_load_2d(const void* gptr, unsigned lds_off,
                                   unsigned t0, unsigned t1, unsigned stride0) {
  unsigned long long ga = (unsigned long long)(uintptr_t)gptr;
  u32x4 g0;
  g0[0] = 1u;                                             // count=1 (user D#)
  g0[1] = lds_off;                                        // LDS byte address
  g0[2] = (unsigned)(ga & 0xFFFFFFFFu);
  g0[3] = (unsigned)((ga >> 32) & 0x01FFFFFFu) | 0x80000000u;  // type=2
  i32x8 g1;
  g1[0] = (int)((1u << 16) | (1u << 20) | (3u << 22) | (3u << 25));
  g1[1] = (int)((t0 & 0xFFFFu) << 16);                    // tensor_dim0 lo16
  g1[2] = (int)(((t0 >> 16) & 0xFFFFu) | ((t1 & 0xFFFFu) << 16));
  g1[3] = (int)(((t1 >> 16) & 0xFFFFu) | ((t0 & 0xFFFFu) << 16)); // tile_dim0
  g1[4] = (int)(t1 & 0xFFFFu);                            // tile_dim1 (tile_dim2=0)
  g1[5] = (int)stride0;                                   // dim0 stride lo32
  g1[6] = 0;
  g1[7] = 0;
  i32x4 gz4 = {0, 0, 0, 0};
  i32x8 gz8 = {0, 0, 0, 0, 0, 0, 0, 0};
  __builtin_amdgcn_tensor_load_to_lds(g0, g1, gz4, gz4, gz8, 0);
}

__device__ inline unsigned lds_addr_of(const void* p) {
  return (unsigned)(uintptr_t)p;   // low 32 bits of generic ptr = LDS byte offset
}

// ---------------- weight / codebook conversion ----------------
// W [co][ci][3][3] f32  ->  Wk [co][(r*3+s)*256 + ci] f16
__global__ __launch_bounds__(256) void k_cvt_w(const float* __restrict__ W,
                                               h_t* __restrict__ Wk) {
  int i = blockIdx.x * 256 + threadIdx.x;
  if (i >= 256 * 2304) return;
  int co = i / 2304, k = i % 2304;
  int tap = k >> 8, ci = k & 255;
  Wk[i] = (h_t)W[((size_t)co * 256 + ci) * 9 + tap];
}

// E f32 -> E16 f16, and E2[row] = 0.5*||e||^2
__global__ __launch_bounds__(256) void k_cvt_E(const float* __restrict__ E,
                                               h_t* __restrict__ E16,
                                               float* __restrict__ E2) {
  int row = blockIdx.x;
  const float* src = E + (size_t)row * DDIM;
  h_t* dst = E16 + (size_t)row * DDIM;
  float s = 0.f;
  for (int e = threadIdx.x; e < DDIM; e += 256) {
    float v = src[e];
    dst[e] = (h_t)v;
    s += v * v;
  }
  __shared__ float red[256];
  red[threadIdx.x] = s;
  __syncthreads();
  for (int off = 128; off > 0; off >>= 1) {
    if (threadIdx.x < off) red[threadIdx.x] += red[threadIdx.x + off];
    __syncthreads();
  }
  if (threadIdx.x == 0) E2[row] = 0.5f * red[0];
}

// ---------------- stable class bucketing (perm = stable argsort(c)) ----------------
__global__ void k_bucket(const int* __restrict__ c, int* __restrict__ perm,
                         int* __restrict__ offs) {
  if (threadIdx.x != 0 || blockIdx.x != 0) return;
  int cnt[NCLS];
  for (int g = 0; g < NCLS; ++g) cnt[g] = 0;
  for (int i = 0; i < N_IMG; ++i) cnt[c[i]]++;
  int run = 0;
  for (int g = 0; g < NCLS; ++g) { offs[g] = run; run += cnt[g]; }
  offs[NCLS] = run;
  int pos[NCLS];
  for (int g = 0; g < NCLS; ++g) pos[g] = offs[g];
  for (int i = 0; i < N_IMG; ++i) { int g = c[i]; perm[pos[g]++] = i; }
}

// ---------------- encoder conv0: 1->256, 28->14, s2 p1, relu, out NHWC f16 ----------------
__global__ __launch_bounds__(256) void k_enc0(const float* __restrict__ x,
                                              const float* __restrict__ W0,
                                              const float* __restrict__ b0,
                                              h_t* __restrict__ out) {
  int n = blockIdx.y;
  int p = blockIdx.x;                 // 0..195
  int oy = p / 14, ox = p % 14;
  int co = threadIdx.x;
  __shared__ float xs[9];
  if (threadIdx.x < 9) {
    int r = threadIdx.x / 3, s = threadIdx.x % 3;
    int iy = oy * 2 - 1 + r, ix = ox * 2 - 1 + s;
    float v = 0.f;
    if (iy >= 0 && iy < 28 && ix >= 0 && ix < 28)
      v = x[((size_t)n * 28 + iy) * 28 + ix];
    xs[threadIdx.x] = v;
  }
  __syncthreads();
  float acc = b0[co];
#pragma unroll
  for (int t = 0; t < 9; ++t) acc += xs[t] * W0[co * 9 + t];
  acc = acc > 0.f ? acc : 0.f;
  out[(((size_t)n * 14 + oy) * 14 + ox) * CD + co] = (h_t)acc;
}

// ---------------- generic 3x3 conv, Cin=Cout=256, implicit-GEMM WMMA ----------------
// Double-buffered TDM pipeline: wave0 prefetches k-step ks+1 while all waves
// run WMMA on ks (TDM ops retire in order per wave -> s_wait_tensorcnt(1)).
// in: NHWC f16 [N][Hin][Win][256]; Wk: [co][k] f16 (k=(r*3+s)*256+ci); bias f32[256]
// mode 0: out NHWC f16 (+optional relu)
// mode 1: enc2 dual write: ze_f32/ze_f16 at [n*4096 + co*16 + (oy*4+ox)] (NCHW-flat)
__global__ __launch_bounds__(256) void k_conv_wmma(
    const h_t* __restrict__ in, h_t* __restrict__ out,
    const h_t* __restrict__ Wk, const float* __restrict__ bias,
    int Hin, int Win, int Hout, int Wout, int stride, int pad,
    int relu, int mode, float* __restrict__ ze_f32, h_t* __restrict__ ze_f16) {
  const int P = Hout * Wout;
  const int n_img = blockIdx.y;
  const int p0 = blockIdx.x * 32;
  const int t = threadIdx.x;
  const int lane = t & 31;
  const int wv = t >> 5;           // 0..7
  const int mw = wv & 1;           // M-subtile
  const int nw = wv >> 1;          // N-group: cols nw*64..+63
  const int halfId = lane >> 4;
  const int ln = lane & 15;
  const int kb = halfId * 8;

  __shared__ h_t As[32][40];
  __shared__ h_t Bs[2][256][40];   // ping-pong TDM dest (32 data + 8 pad halves/row)

  const unsigned bs_lds[2] = {lds_addr_of(&Bs[0][0][0]), lds_addr_of(&Bs[1][0][0])};

  v8f acc[4];
#pragma unroll
  for (int j = 0; j < 4; ++j) acc[j] = (v8f){};

  const int ap = t >> 3;           // pixel within tile 0..31
  const int ac0 = (t & 7) * 4;     // ci offset within 32-slice
  const int Pg_a = p0 + ap;
  const int oy = Pg_a / Wout, ox = Pg_a % Wout;

  if (wv == 0)                     // prologue: DMA slab for ks=0 into buf0
    tdm_load_2d(Wk, bs_lds[0], 32u, 256u, 2304u);

  for (int ks = 0; ks < 72; ++ks) {
    const int kpos = ks * 32;
    const int tap = kpos >> 8;     // 0..8
    const int ci0 = kpos & 255;
    const int r = tap / 3, s = tap % 3;
    const int cur = ks & 1;

    h4 av = (h4){};
    const int iy = oy * stride - pad + r;
    const int ix = ox * stride - pad + s;
    if (Pg_a < P && iy >= 0 && iy < Hin && ix >= 0 && ix < Win)
      av = *(const h4*)(in + (((size_t)n_img * Hin + iy) * Win + ix) * CD + ci0 + ac0);

    __syncthreads();               // reads of As & Bs[cur^1] (step ks-1) done
    *(h4*)&As[ap][ac0] = av;
    if (wv == 0) {
      if (ks + 1 < 72) {           // prefetch next slab into the idle buffer
        tdm_load_2d(Wk + kpos + 32, bs_lds[cur ^ 1], 32u, 256u, 2304u);
        __builtin_amdgcn_s_wait_tensorcnt(1);   // current slab landed; next in flight
      } else {
        __builtin_amdgcn_s_wait_tensorcnt(0);
      }
    }
    __syncthreads();               // publish As + Bs[cur]

    const v16h a = frag_ld(&As[mw * 16 + ln][0], kb);
#pragma unroll
    for (int j = 0; j < 4; ++j) {
      const v16h b = frag_ld(&Bs[cur][nw * 64 + j * 16 + ln][0], kb);
      acc[j] = wmma16(a, b, acc[j]);
    }
  }

  // epilogue: C layout — VGPR rr: lanes0-15 M=rr, lanes16-31 M=rr+8; N=lane&15
#pragma unroll
  for (int j = 0; j < 4; ++j) {
    const int co = nw * 64 + j * 16 + ln;
    const float bv = bias[co];
#pragma unroll
    for (int rr = 0; rr < 8; ++rr) {
      const int m = mw * 16 + halfId * 8 + rr;
      const int Pg = p0 + m;
      if (Pg >= P) continue;
      float v = acc[j][rr] + bv;
      if (relu) v = v > 0.f ? v : 0.f;
      if (mode == 0) {
        out[((size_t)n_img * P + Pg) * CD + co] = (h_t)v;
      } else {
        const size_t zi = (size_t)n_img * DDIM + (size_t)co * 16 + Pg;
        ze_f32[zi] = v;
        ze_f16[zi] = (h_t)v;
      }
    }
  }
}

// ---------------- VQ: per-class WMMA GEMM, score = z.e - 0.5||e||^2 ----------------
// grid: (x = 16 code-tiles of 128, y = 32 sample-tiles of 16, z = 8 classes)
__global__ __launch_bounds__(256) void k_vq_scores(
    const h_t* __restrict__ ze16, const h_t* __restrict__ E16,
    const float* __restrict__ E2, const int* __restrict__ perm,
    const int* __restrict__ offs, float* __restrict__ partS,
    int* __restrict__ partI) {
  const int g = blockIdx.z;
  const int m0 = blockIdx.y * 16;
  const int nt = blockIdx.x;
  const int cnt = offs[g + 1] - offs[g];
  if (m0 >= cnt) return;
  const int c0 = nt * 128;
  const int t = threadIdx.x;
  const int lane = t & 31, wv = t >> 5;
  const int halfId = lane >> 4, ln = lane & 15;
  const int kb = halfId * 8;

  __shared__ h_t As[16][40];
  __shared__ h_t Bs[2][128][40];   // ping-pong TDM dest
  __shared__ float scoreS[16][128];
  __shared__ int samp_s[16];

  const unsigned bs_lds[2] = {lds_addr_of(&Bs[0][0][0]), lds_addr_of(&Bs[1][0][0])};

  if (t < 16) {
    int mm = m0 + t;
    if (mm >= cnt) mm = cnt - 1;   // pad rows duplicate last (results discarded)
    samp_s[t] = perm[offs[g] + mm];
  }
  __syncthreads();

  const int am = t >> 4;
  const int ak = (t & 15) * 2;
  const size_t arow = (size_t)samp_s[am] * DDIM;
  const h_t* Ebase = E16 + ((size_t)g * KCODE + c0) * DDIM;

  if (wv == 0)                     // prologue: slab for k0=0
    tdm_load_2d(Ebase, bs_lds[0], 32u, 128u, 4096u);

  v8f acc = (v8f){};
  for (int ks = 0; ks < 128; ++ks) {
    const int k0 = ks * 32;
    const int cur = ks & 1;
    __syncthreads();
    *(h2*)&As[am][ak] = *(const h2*)(ze16 + arow + k0 + ak);
    if (wv == 0) {
      if (ks + 1 < 128) {
        tdm_load_2d(Ebase + k0 + 32, bs_lds[cur ^ 1], 32u, 128u, 4096u);
        __builtin_amdgcn_s_wait_tensorcnt(1);
      } else {
        __builtin_amdgcn_s_wait_tensorcnt(0);
      }
    }
    __syncthreads();
    const v16h a = frag_ld(&As[ln][0], kb);
    const v16h b = frag_ld(&Bs[cur][wv * 16 + ln][0], kb);
    acc = wmma16(a, b, acc);
  }

  {
    const int code = c0 + wv * 16 + ln;
    const float e2 = E2[(size_t)g * KCODE + code];
#pragma unroll
    for (int rr = 0; rr < 8; ++rr) {
      const int m = halfId * 8 + rr;
      scoreS[m][wv * 16 + ln] = acc[rr] - e2;
    }
  }
  __syncthreads();

  if (t < 16 && (m0 + t) < cnt) {
    float best = scoreS[t][0];
    int bi = 0;
    for (int n = 1; n < 128; ++n) {
      float v = scoreS[t][n];
      if (v > best) { best = v; bi = n; }   // strict > keeps earliest (argmin tie rule)
    }
    const int p = offs[g] + m0 + t;         // perm-position (class-sorted order)
    partS[p * 16 + nt] = best;
    partI[p * 16 + nt] = g * KCODE + c0 + bi;
  }
}

__global__ void k_vq_reduce(const float* __restrict__ partS,
                            const int* __restrict__ partI,
                            int* __restrict__ gidx) {
  int i = blockIdx.x * blockDim.x + threadIdx.x;
  if (i >= N_IMG) return;
  float best = partS[i * 16];
  int bi = partI[i * 16];
  for (int t = 1; t < 16; ++t) {
    float v = partS[i * 16 + t];
    if (v > best) { best = v; bi = partI[i * 16 + t]; }
  }
  gidx[i] = bi;
}

// z_q_bar_p (f32 output) and decoder input (f16) = E[idx], in perm order
__global__ __launch_bounds__(256) void k_gather(const float* __restrict__ E,
                                                const int* __restrict__ gidx,
                                                float* __restrict__ zq_out,
                                                h_t* __restrict__ zq16) {
  const int i = blockIdx.x;
  const float* src = E + (size_t)gidx[i] * DDIM;
  for (int e = threadIdx.x; e < DDIM; e += 256) {
    float v = src[e];
    zq_out[(size_t)i * DDIM + e] = v;
    zq16[(size_t)i * DDIM + e] = (h_t)v;
  }
}

// ---------------- bilinear x2 upsample (align_corners=True), out NHWC f16 ----------------
__global__ __launch_bounds__(256) void k_upsample(const h_t* __restrict__ in,
                                                  h_t* __restrict__ out,
                                                  int H, int W, int nchw_in) {
  const int Ho = 2 * H, Wo2 = 2 * W;
  const int n = blockIdx.y;
  const int p = blockIdx.x;
  const int yo = p / Wo2, xo = p % Wo2;
  const float fy = (float)yo * (float)(H - 1) / (float)(Ho - 1);
  const float fx = (float)xo * (float)(W - 1) / (float)(Wo2 - 1);
  int y0 = (int)floorf(fy); int y1 = y0 + 1 < H ? y0 + 1 : H - 1;
  int x0 = (int)floorf(fx); int x1 = x0 + 1 < W ? x0 + 1 : W - 1;
  const float wy = fy - (float)y0, wx = fx - (float)x0;
  const int cch = threadIdx.x;
  float v00, v01, v10, v11;
  if (nchw_in) {  // only used for the 4x4 latent: idx = n*4096 + c*16 + y*4 + x
    const size_t base = (size_t)n * DDIM + (size_t)cch * 16;
    v00 = (float)in[base + y0 * 4 + x0];
    v01 = (float)in[base + y0 * 4 + x1];
    v10 = (float)in[base + y1 * 4 + x0];
    v11 = (float)in[base + y1 * 4 + x1];
  } else {
    const size_t rb = (size_t)n * H;
    v00 = (float)in[((rb + y0) * W + x0) * CD + cch];
    v01 = (float)in[((rb + y0) * W + x1) * CD + cch];
    v10 = (float)in[((rb + y1) * W + x0) * CD + cch];
    v11 = (float)in[((rb + y1) * W + x1) * CD + cch];
  }
  const float v = (1.f - wy) * ((1.f - wx) * v00 + wx * v01) +
                  wy * ((1.f - wx) * v10 + wx * v11);
  out[(((size_t)n * Ho + yo) * Wo2 + xo) * CD + cch] = (h_t)v;
}

// ---------------- output conv 256->1, 3x3 p1, sigmoid ----------------
__global__ __launch_bounds__(256) void k_outconv(const h_t* __restrict__ in,
                                                 const float* __restrict__ Wo,
                                                 const float* __restrict__ bo,
                                                 float* __restrict__ out) {
  const int n = blockIdx.y, p = blockIdx.x;
  const int oy = p / 28, ox = p % 28;
  const int ci = threadIdx.x;
  float acc = 0.f;
#pragma unroll
  for (int tap = 0; tap < 9; ++tap) {
    const int r = tap / 3, s = tap % 3;
    const int iy = oy - 1 + r, ix = ox - 1 + s;
    if (iy < 0 || iy >= 28 || ix < 0 || ix >= 28) continue;
    acc += (float)in[(((size_t)n * 28 + iy) * 28 + ix) * CD + ci] * Wo[ci * 9 + tap];
  }
  __shared__ float red[256];
  red[ci] = acc;
  __syncthreads();
  for (int off = 128; off > 0; off >>= 1) {
    if (ci < off) red[ci] += red[ci + off];
    __syncthreads();
  }
  if (ci == 0) {
    const float v = red[0] + bo[0];
    out[(size_t)n * 784 + p] = 1.f / (1.f + expf(-v));
  }
}

// ---------------- launch ----------------
extern "C" void kernel_launch(void* const* d_in, const int* in_sizes, int n_in,
                              void* d_out, int out_size, void* d_ws, size_t ws_size,
                              hipStream_t stream) {
  (void)in_sizes; (void)n_in; (void)out_size; (void)ws_size;
  const float* x   = (const float*)d_in[0];
  const float* We0 = (const float*)d_in[1];
  const float* be0 = (const float*)d_in[2];
  const float* We1 = (const float*)d_in[3];
  const float* be1 = (const float*)d_in[4];
  const float* We2 = (const float*)d_in[5];
  const float* be2 = (const float*)d_in[6];
  const float* E   = (const float*)d_in[7];
  const float* Wd0 = (const float*)d_in[8];
  const float* bd0 = (const float*)d_in[9];
  const float* Wd1 = (const float*)d_in[10];
  const float* bd1 = (const float*)d_in[11];
  const float* Wd2 = (const float*)d_in[12];
  const float* bd2 = (const float*)d_in[13];
  const float* Wo  = (const float*)d_in[14];
  const float* bo  = (const float*)d_in[15];
  const int*   c   = (const int*)d_in[16];

  float* out_x  = (float*)d_out;                        // [512,1,28,28]
  float* out_ze = out_x + (size_t)N_IMG * 784;          // [512,4096,1,1]
  float* out_zq = out_ze + (size_t)N_IMG * DDIM;        // [512,4096,1,1]

  char* ws = (char*)d_ws;
  size_t off = 0;
  auto take = [&](size_t bytes) -> char* {
    off = (off + 255) & ~(size_t)255;
    char* p = ws + off;
    off += bytes;
    return p;
  };
  const size_t WKB = (size_t)2304 * 256 * sizeof(h_t);  // one layer's B matrix
  h_t*   Wk    = (h_t*)take(5 * WKB);                   // We1,We2,Wd0,Wd1,Wd2
  h_t*   E16   = (h_t*)take((size_t)KCODE * NCLS * DDIM * sizeof(h_t));
  float* E2    = (float*)take((size_t)KCODE * NCLS * sizeof(float));
  h_t*   ze16  = (h_t*)take((size_t)N_IMG * DDIM * sizeof(h_t));
  h_t*   zq16  = (h_t*)take((size_t)N_IMG * DDIM * sizeof(h_t));
  int*   perm  = (int*)take(N_IMG * sizeof(int));
  int*   offs  = (int*)take((NCLS + 1) * sizeof(int));
  float* partS = (float*)take((size_t)N_IMG * 16 * sizeof(float));
  int*   partI = (int*)take((size_t)N_IMG * 16 * sizeof(int));
  int*   gidx  = (int*)take(N_IMG * sizeof(int));
  h_t*   bufA  = (h_t*)take((size_t)N_IMG * 28 * 28 * CD * sizeof(h_t));
  h_t*   bufB  = (h_t*)take((size_t)N_IMG * 28 * 28 * CD * sizeof(h_t));

  const float* wsrc[5] = {We1, We2, Wd0, Wd1, Wd2};
  for (int l = 0; l < 5; ++l)
    k_cvt_w<<<dim3((256 * 2304 + 255) / 256), 256, 0, stream>>>(
        wsrc[l], Wk + (size_t)l * 2304 * 256);
  k_cvt_E<<<dim3(KCODE * NCLS), 256, 0, stream>>>(E, E16, E2);
  k_bucket<<<dim3(1), 32, 0, stream>>>(c, perm, offs);

  // encoder
  k_enc0<<<dim3(196, N_IMG), 256, 0, stream>>>(x, We0, be0, bufA);
  k_conv_wmma<<<dim3(2, N_IMG), 256, 0, stream>>>(                 // 14->7, relu
      bufA, bufB, Wk + 0 * 2304 * 256, be1, 14, 14, 7, 7, 2, 1, 1, 0,
      nullptr, nullptr);
  k_conv_wmma<<<dim3(1, N_IMG), 256, 0, stream>>>(                 // 7->4, z_e
      bufB, nullptr, Wk + 1 * (size_t)2304 * 256, be2, 7, 7, 4, 4, 2, 1, 0, 1,
      out_ze, ze16);

  // class-conditional VQ
  k_vq_scores<<<dim3(16, 32, NCLS), 256, 0, stream>>>(ze16, E16, E2, perm, offs,
                                                      partS, partI);
  k_vq_reduce<<<dim3(2), 256, 0, stream>>>(partS, partI, gidx);
  k_gather<<<dim3(N_IMG), 256, 0, stream>>>(E, gidx, out_zq, zq16);

  // decoder (operates in perm order; x_tilde is returned permuted, matching ref)
  k_upsample<<<dim3(64, N_IMG), 256, 0, stream>>>(zq16, bufA, 4, 4, 1);   // 4->8
  k_conv_wmma<<<dim3(2, N_IMG), 256, 0, stream>>>(                        // 8x8 p1
      bufA, bufB, Wk + 2 * (size_t)2304 * 256, bd0, 8, 8, 8, 8, 1, 1, 1, 0,
      nullptr, nullptr);
  k_upsample<<<dim3(256, N_IMG), 256, 0, stream>>>(bufB, bufA, 8, 8, 0);  // 8->16
  k_conv_wmma<<<dim3(7, N_IMG), 256, 0, stream>>>(                        // 16->14 p0
      bufA, bufB, Wk + 3 * (size_t)2304 * 256, bd1, 16, 16, 14, 14, 1, 0, 1, 0,
      nullptr, nullptr);
  k_upsample<<<dim3(784, N_IMG), 256, 0, stream>>>(bufB, bufA, 14, 14, 0); // 14->28
  k_conv_wmma<<<dim3(25, N_IMG), 256, 0, stream>>>(                       // 28x28 p1
      bufA, bufB, Wk + 4 * (size_t)2304 * 256, bd2, 28, 28, 28, 28, 1, 1, 1, 0,
      nullptr, nullptr);
  k_outconv<<<dim3(784, N_IMG), 256, 0, stream>>>(bufB, Wo, bo, out_x);
}
